// FluxAttention_28853590294598
// MI455X (gfx1250) — compile-verified, matrix operand order checked by reading, not run on hardware
//
#include <hip/hip_runtime.h>

// ---------------------------------------------------------------------------
// CDNA5 (gfx1250) Flux attention block: bf16 WMMA GEMMs + flash attention.
// wave32. Fragment loads use the table-order 16-bit A-layout pattern
// (2 x ds_load_b128); operands needing the B layout are staged transposed.
// Flash attention K/V tiles are DMA'd into LDS by the Tensor Data Mover
// (tensor_load_to_lds + s_wait_tensorcnt) with ping-pong double buffering.
// ---------------------------------------------------------------------------

typedef __attribute__((ext_vector_type(16))) __bf16 v16bf;
typedef __attribute__((ext_vector_type(8)))  float  v8f;
typedef __attribute__((ext_vector_type(4)))  unsigned int u32x4;
typedef __attribute__((ext_vector_type(8)))  int i32x8;
typedef __attribute__((ext_vector_type(4)))  int i32x4;

#if defined(__gfx1250__) && __has_builtin(__builtin_amdgcn_tensor_load_to_lds) && \
    __has_builtin(__builtin_amdgcn_s_wait_tensorcnt)
#define USE_TDM 1
#else
#define USE_TDM 0
#endif

#define WMMA_BF16(a, b, c) \
  __builtin_amdgcn_wmma_f32_16x16x32_bf16(false, (a), false, (b), (short)0, (c), false, false)

// 16-bit 16x32 A-layout gather (ISA 7.12.2): lane 0-15 -> M=lane, K-half 0;
// lane 16-31 -> K-half 8. elems 0..7 = K kh+0..7, elems 8..15 = K kh+16..23.
__device__ __forceinline__ v16bf frag_a_lds(const __bf16* base, int stride) {
  const int lane = threadIdx.x & 31;
  const __bf16* p = base + (lane & 15) * stride + ((lane >> 4) << 3);
  v16bf a;
#pragma unroll
  for (int e = 0; e < 8; ++e) { a[e] = p[e]; a[8 + e] = p[16 + e]; }
  return a;
}

#if USE_TDM
// TDM 2D tile load: Global -> LDS, bf16 elements (data_size=2B), with LDS row
// padding (pad_interval/pad_amount in DWORDs per ISA ch8). Groups 2/3 zero
// (<=2D tensor). Issued once per wave; tracked with TENSORcnt.
// 6-arg builtin form: (u32x4 g0, i32x8 g1, i32x4 g2, i32x4 g3, i32x8, i32 cpol).
__device__ __forceinline__ void tdm_load_2d(unsigned lds_addr, unsigned long long ga,
                                            unsigned tile_d0, unsigned tile_d1,
                                            unsigned tensor_d0, unsigned tensor_d1,
                                            unsigned long long stride_d0,
                                            unsigned pad_interval, unsigned pad_amount) {
  u32x4 g0 = {0u, 0u, 0u, 0u};
  g0[0] = 1u;                                             // count=1, user mode
  g0[1] = lds_addr;                                       // LDS byte address
  g0[2] = (unsigned)(ga & 0xFFFFFFFFull);                 // global_addr[31:0]
  g0[3] = (unsigned)((ga >> 32) & 0x01FFFFFFull) | (2u << 30);  // addr[56:32] | type=2
  i32x8 g1 = {0, 0, 0, 0, 0, 0, 0, 0};
  g1[0] = (int)((1u << 16) | (1u << 20) |                 // data_size=2B, pad_enable
                (pad_interval << 22) | (pad_amount << 25));
  g1[1] = (int)((tensor_d0 & 0xFFFFu) << 16);             // tensor_dim0 lo
  g1[2] = (int)(((tensor_d0 >> 16) & 0xFFFFu) | ((tensor_d1 & 0xFFFFu) << 16));
  g1[3] = (int)(((tensor_d1 >> 16) & 0xFFFFu) | ((tile_d0 & 0xFFFFu) << 16));
  g1[4] = (int)(tile_d1 & 0xFFFFu);                       // tile_dim1 (tile_dim2=0)
  g1[5] = (int)(stride_d0 & 0xFFFFFFFFull);               // tensor_dim0_stride lo
  g1[6] = (int)((stride_d0 >> 32) & 0xFFFFull);           // stride hi (dim1_stride=0)
  i32x4 z4 = {0, 0, 0, 0};
  i32x8 z8 = {0, 0, 0, 0, 0, 0, 0, 0};
  __builtin_amdgcn_tensor_load_to_lds(g0, g1, z4, z4, z8, 0);
}
#endif

// ---------------------------------------------------------------------------
// Staging helpers (256-thread blocks, 128x32 A tile, 32x128 B tile stored
// transposed as sBt[n][k] so B fragments use the same A-gather).
// ---------------------------------------------------------------------------
__device__ __forceinline__ void stage_a_f32(const float* __restrict__ X,
                                            __bf16 (*sA)[40],
                                            int m0, int K, int k0, int t) {
#pragma unroll
  for (int jj = 0; jj < 4; ++jj) {
    int lin = t + 256 * jj;
    int row = lin >> 3, kq = lin & 7;
    float4 va = *(const float4*)(X + (size_t)(m0 + row) * K + k0 + kq * 4);
    __bf16* d = &sA[row][kq * 4];
    d[0] = (__bf16)va.x; d[1] = (__bf16)va.y; d[2] = (__bf16)va.z; d[3] = (__bf16)va.w;
  }
}

__device__ __forceinline__ void stage_a_bf16(const __bf16* __restrict__ A,
                                             __bf16 (*sA)[40],
                                             int m0, int K, int k0, int t) {
#pragma unroll
  for (int jj = 0; jj < 4; ++jj) {
    int lin = t + 256 * jj;
    int m = lin >> 3, kq = lin & 7;
    *(uint2*)&sA[m][kq * 4] = *(const uint2*)(A + (size_t)(m0 + m) * K + k0 + kq * 4);
  }
}

__device__ __forceinline__ void stage_bt_f32(const float* __restrict__ W,
                                             __bf16 (*sBt)[40],
                                             int n0, int N, int k0, int t) {
  const int sn = t & 127;
  const int sk = (t >> 7) * 16;
  float tf[16];
#pragma unroll
  for (int i = 0; i < 16; ++i) tf[i] = W[(size_t)(k0 + sk + i) * N + n0 + sn];
  __bf16* d = &sBt[sn][sk];
#pragma unroll
  for (int i = 0; i < 16; ++i) d[i] = (__bf16)tf[i];
}

// ---------------------------------------------------------------------------
// GEMM 1: Y(bf16, row_off..) = X(f32, MxK) @ W(f32, KxN) -> bf16.
// 128x128 block tile, 8 waves of 64x32 (4x2 WMMA tiles), K step 32,
// ping-pong LDS double buffering (one barrier per K-step).
// ---------------------------------------------------------------------------
__global__ __launch_bounds__(256)
void gemm_x_w_bf16out(const float* __restrict__ X, const float* __restrict__ W,
                      unsigned short* __restrict__ Yus,
                      int M, int N, int K, int row_off) {
  __bf16* Y = (__bf16*)Yus;
  __shared__ __bf16 sA[2][128][40];
  __shared__ __bf16 sBt[2][128][40];

  const int m0 = blockIdx.y * 128;
  const int n0 = blockIdx.x * 128;
  const int t  = threadIdx.x;
  const int wave = t >> 5;
  const int wm = (wave & 1) * 64;
  const int wn = (wave >> 1) * 32;

  const v8f zf = {0.f,0.f,0.f,0.f,0.f,0.f,0.f,0.f};
  v8f acc[4][2];
#pragma unroll
  for (int i = 0; i < 4; ++i)
#pragma unroll
    for (int j = 0; j < 2; ++j) acc[i][j] = zf;

  stage_a_f32(X, sA[0], m0, K, 0, t);
  stage_bt_f32(W, sBt[0], n0, N, 0, t);
  __syncthreads();

  const int nk = K >> 5;
  for (int ks = 0; ks < nk; ++ks) {
    const int cur = ks & 1;
    if (ks + 1 < nk) {
      if (ks + 2 < nk)  // speculative prefetch two tiles ahead
        __builtin_prefetch(W + (size_t)((ks + 2) * 32) * N + n0 + (t & 127), 0, 1);
      stage_a_f32(X, sA[cur ^ 1], m0, K, (ks + 1) * 32, t);
      stage_bt_f32(W, sBt[cur ^ 1], n0, N, (ks + 1) * 32, t);
    }
    v16bf af[4], bfr[2];
#pragma unroll
    for (int i = 0; i < 4; ++i) af[i] = frag_a_lds(&sA[cur][wm + i * 16][0], 40);
#pragma unroll
    for (int j = 0; j < 2; ++j) bfr[j] = frag_a_lds(&sBt[cur][wn + j * 16][0], 40);
#pragma unroll
    for (int i = 0; i < 4; ++i)
#pragma unroll
      for (int j = 0; j < 2; ++j)
        acc[i][j] = WMMA_BF16(af[i], bfr[j], acc[i][j]);
    __syncthreads();
  }

  const int lane = t & 31;
  const int col  = lane & 15;
  const int rb   = (lane >> 4) * 8;
#pragma unroll
  for (int i = 0; i < 4; ++i)
#pragma unroll
    for (int j = 0; j < 2; ++j)
#pragma unroll
      for (int r = 0; r < 8; ++r) {
        size_t grow = (size_t)(row_off + m0 + wm + i * 16 + rb + r);
        Y[grow * N + (n0 + wn + j * 16 + col)] = (__bf16)acc[i][j][r];
      }
}

// ---------------------------------------------------------------------------
// RMSNorm(q,k) + RoPE + head-major scatter (Q, K only).
// One wave per (token p, head h); lane owns 4 d.
// ---------------------------------------------------------------------------
__global__ __launch_bounds__(256)
void norm_rope(const unsigned short* __restrict__ qkvus, const float* __restrict__ rot,
               const float* __restrict__ qn, const float* __restrict__ kn,
               const float* __restrict__ eqn, const float* __restrict__ ekn,
               unsigned short* __restrict__ Qus, unsigned short* __restrict__ Kus) {
  const __bf16* qkv = (const __bf16*)qkvus;
  __bf16* Qh = (__bf16*)Qus;
  __bf16* Kh = (__bf16*)Kus;

  const int wave = threadIdx.x >> 5, lane = threadIdx.x & 31;
  const int rid = blockIdx.x * 8 + wave;       // 0 .. 4608*24-1
  const int p = rid / 24, h = rid % 24;
  const int d0 = lane * 4;

  const float* qsc = (p < 512) ? eqn : qn;
  const float* ksc = (p < 512) ? ekn : kn;

  const __bf16* row = qkv + (size_t)p * 9216 + h * 128 + d0;
  float q[4], k[4];
#pragma unroll
  for (int j = 0; j < 4; ++j) {
    q[j] = (float)row[j];
    k[j] = (float)row[3072 + j];
  }
  float sq = 0.f, sk = 0.f;
#pragma unroll
  for (int j = 0; j < 4; ++j) { sq += q[j] * q[j]; sk += k[j] * k[j]; }
#pragma unroll
  for (int off = 16; off; off >>= 1) {
    sq += __shfl_xor(sq, off, 32);
    sk += __shfl_xor(sk, off, 32);
  }
  const float rq = rsqrtf(sq * (1.f / 128.f) + 1e-6f);
  const float rk = rsqrtf(sk * (1.f / 128.f) + 1e-6f);
#pragma unroll
  for (int j = 0; j < 4; ++j) { q[j] *= rq * qsc[d0 + j]; k[j] *= rk * ksc[d0 + j]; }

  // RoPE: pair i uses fc[p][i][j][c] = rot[p*256 + i*4 + j*2 + c]
  const float* fr = rot + (size_t)p * 256 + lane * 8;  // two pairs per lane
  float qo[4], ko[4];
  qo[0] = fr[0] * q[0] + fr[1] * q[1];  qo[1] = fr[2] * q[0] + fr[3] * q[1];
  qo[2] = fr[4] * q[2] + fr[5] * q[3];  qo[3] = fr[6] * q[2] + fr[7] * q[3];
  ko[0] = fr[0] * k[0] + fr[1] * k[1];  ko[1] = fr[2] * k[0] + fr[3] * k[1];
  ko[2] = fr[4] * k[2] + fr[5] * k[3];  ko[3] = fr[6] * k[2] + fr[7] * k[3];

  const float qscale = 0.08838834764831845f;  // 128^-0.5 folded into Q
  const size_t ob = ((size_t)h * 4608 + p) * 128 + d0;
#pragma unroll
  for (int j = 0; j < 4; ++j) {
    Qh[ob + j] = (__bf16)(qo[j] * qscale);
    Kh[ob + j] = (__bf16)ko[j];
  }
}

// ---------------------------------------------------------------------------
// V transpose: qkv_raw V part (p-major) -> Vt[h][d][p] (d-major). LDS-tiled,
// coalesced on both sides. grid = (24 heads, 72 p-tiles, 2 d-tiles).
// ---------------------------------------------------------------------------
__global__ __launch_bounds__(256)
void transpose_v(const unsigned short* __restrict__ qkvus,
                 unsigned short* __restrict__ Vtus) {
  const __bf16* qkv = (const __bf16*)qkvus;
  __bf16* Vt = (__bf16*)Vtus;
  __shared__ __bf16 tile[64][76];

  const int h  = blockIdx.x;
  const int pt = blockIdx.y;
  const int dt = blockIdx.z;
  const int t  = threadIdx.x;

#pragma unroll
  for (int i = 0; i < 16; ++i) {
    int lin = t + 256 * i;
    int pr = lin >> 6, dc = lin & 63;
    tile[pr][dc] = qkv[(size_t)(pt * 64 + pr) * 9216 + 6144 + h * 128 + dt * 64 + dc];
  }
  __syncthreads();
#pragma unroll
  for (int i = 0; i < 16; ++i) {
    int lin = t + 256 * i;
    int dr = lin >> 6, pc = lin & 63;
    Vt[(size_t)(h * 128 + dt * 64 + dr) * 4608 + pt * 64 + pc] = tile[pc][dr];
  }
}

// ---------------------------------------------------------------------------
// Flash attention: block = (head h, 64 Q rows), 4 waves x 16 rows.
// Double-buffered 64-key tiles; TDM DMA (tensor_load_to_lds) stages the next
// K/V tile while WMMAs consume the current one.
// ---------------------------------------------------------------------------
__global__ __launch_bounds__(128)
void flash_attn(const unsigned short* __restrict__ Qus,
                const unsigned short* __restrict__ Kus,
                const unsigned short* __restrict__ Vtus,
                unsigned short* __restrict__ Ous) {
  const __bf16* Qh = (const __bf16*)Qus;
  const __bf16* Kh = (const __bf16*)Kus;
  const __bf16* Vt = (const __bf16*)Vtus;
  __bf16* Oout = (__bf16*)Ous;

  __shared__ __bf16 sK[2][64][136];    // keys x d (64 DW row + 4 DW pad)
  __shared__ __bf16 sVt[2][128][72];   // d x keys (32 DW row + 4 DW pad)
  __shared__ float  sS[4][16][68];
  __shared__ float  sM[4][16];
  __shared__ float  sL[4][16];
  __shared__ float  sAl[4][16];

  const int h    = blockIdx.y;
  const int qt   = blockIdx.x;
  const int wave = threadIdx.x >> 5;
  const int lane = threadIdx.x & 31;
  const int wr0  = wave * 16;

  const __bf16* Qb = Qh + ((size_t)h * 4608 + qt * 64) * 128;

  // Q fragments (16 rows x 128 d = 4 A-frags), loaded once
  v16bf qf[4];
#pragma unroll
  for (int kc = 0; kc < 4; ++kc) {
    const __bf16* p = Qb + (size_t)(wr0 + (lane & 15)) * 128 + kc * 32 + ((lane >> 4) << 3);
    v16bf a;
#pragma unroll
    for (int e = 0; e < 8; ++e) { a[e] = p[e]; a[8 + e] = p[16 + e]; }
    qf[kc] = a;
  }

  const v8f zf = {0.f,0.f,0.f,0.f,0.f,0.f,0.f,0.f};
  v8f o[8];
#pragma unroll
  for (int tt = 0; tt < 8; ++tt) o[tt] = zf;

  if (lane < 16) { sM[wave][lane] = -3.0e38f; sL[wave][lane] = 0.f; }

#if USE_TDM
  // Wave 0 drives the Tensor Data Mover; other waves sync on barriers.
  auto issue_tiles = [&](int kt2, int buf) {
    if (threadIdx.x < 32) {
      unsigned ldsK = (unsigned)(unsigned long long)(void*)&sK[buf][0][0];
      unsigned ldsV = (unsigned)(unsigned long long)(void*)&sVt[buf][0][0];
      unsigned long long gaK =
          (unsigned long long)(const void*)(Kh + ((size_t)h * 4608 + kt2 * 64) * 128);
      unsigned long long gaV =
          (unsigned long long)(const void*)(Vt + (size_t)h * 128 * 4608 + kt2 * 64);
      // K tile: 64 rows x 128 bf16; LDS row 64 DW + 4 DW pad (interval 2^(5+1)=64)
      tdm_load_2d(ldsK, gaK, 128u, 64u, 128u, 4608u, 128ull, 5u, 3u);
      // Vt tile: 128 rows x 64 bf16; LDS row 32 DW + 4 DW pad (interval 2^(4+1)=32)
      tdm_load_2d(ldsV, gaV, 64u, 128u, 4608u, 128u, 4608ull, 4u, 3u);
    }
  };
#else
  auto copy_tiles = [&](int kt2, int buf) {
    const int r = threadIdx.x >> 1;
    const int half = threadIdx.x & 1;
    const uint4* gk = (const uint4*)(Kh + ((size_t)h * 4608 + kt2 * 64 + r) * 128 + half * 64);
    uint4* dk = (uint4*)(&sK[buf][r][half * 64]);
#pragma unroll
    for (int i = 0; i < 8; ++i) dk[i] = gk[i];
    const uint4* gv = (const uint4*)(Vt + ((size_t)h * 128 + threadIdx.x) * 4608 + kt2 * 64);
    uint4* dv = (uint4*)(&sVt[buf][threadIdx.x][0]);
#pragma unroll
    for (int i = 0; i < 8; ++i) dv[i] = gv[i];
  };
#endif

#if USE_TDM
  issue_tiles(0, 0);
  if (threadIdx.x < 32) __builtin_amdgcn_s_wait_tensorcnt((short)0);
#else
  copy_tiles(0, 0);
#endif
  __syncthreads();

  const int rb = (lane >> 4) * 8;

  for (int kt = 0; kt < 72; ++kt) {
    const int cur = kt & 1;
    if (kt + 1 < 72) {
#if USE_TDM
      issue_tiles(kt + 1, cur ^ 1);   // DMA next tile while computing this one
#else
      copy_tiles(kt + 1, cur ^ 1);
#endif
    }

    // ---- S = Q K^T (scale folded into Q) ----
#pragma unroll
    for (int n = 0; n < 4; ++n) {
      v8f s = zf;
#pragma unroll
      for (int kc = 0; kc < 4; ++kc) {
        v16bf bfr = frag_a_lds(&sK[cur][n * 16][kc * 32], 136);  // = K^T B-fragment
        s = WMMA_BF16(qf[kc], bfr, s);
      }
      const int col = lane & 15;
#pragma unroll
      for (int r = 0; r < 8; ++r) sS[wave][rb + r][n * 16 + col] = s[r];
    }
    __syncthreads();

    // ---- online softmax (lanes 0-15 each own one Q row; float4 scans) ----
    if (lane < 16) {
      float mo = sM[wave][lane], lo = sL[wave][lane];
      float mx = mo;
      float4* rp = (float4*)&sS[wave][lane][0];
#pragma unroll 4
      for (int j = 0; j < 16; ++j) {
        float4 x = rp[j];
        mx = fmaxf(mx, fmaxf(fmaxf(x.x, x.y), fmaxf(x.z, x.w)));
      }
      const float alpha = __expf(mo - mx);
      float sum = 0.f;
#pragma unroll 4
      for (int j = 0; j < 16; ++j) {
        float4 x = rp[j];
        x.x = __expf(x.x - mx); x.y = __expf(x.y - mx);
        x.z = __expf(x.z - mx); x.w = __expf(x.w - mx);
        sum += (x.x + x.y) + (x.z + x.w);
        rp[j] = x;
      }
      sM[wave][lane] = mx;
      sL[wave][lane] = lo * alpha + sum;
      sAl[wave][lane] = alpha;
    }
    __syncthreads();

    // ---- rescale O, then O += P V ----
    float al[8];
#pragma unroll
    for (int r = 0; r < 8; ++r) al[r] = sAl[wave][rb + r];
#pragma unroll
    for (int tt = 0; tt < 8; ++tt)
#pragma unroll
      for (int r = 0; r < 8; ++r) o[tt][r] *= al[r];

#pragma unroll
    for (int kc2 = 0; kc2 < 2; ++kc2) {
      // P A-frag: rows 0..15, K = kc2*32.. (f32 -> bf16)
      v16bf pf;
      {
        const float* pr = &sS[wave][lane & 15][kc2 * 32 + ((lane >> 4) << 3)];
#pragma unroll
        for (int e = 0; e < 8; ++e) { pf[e] = (__bf16)pr[e]; pf[8 + e] = (__bf16)pr[16 + e]; }
      }
#pragma unroll
      for (int tt = 0; tt < 8; ++tt) {
        // B[k][n'] = V[key k][d] = sVt[tt*16+n'][kc2*32+k] => A-gather on sVt
        v16bf vf = frag_a_lds(&sVt[cur][tt * 16][kc2 * 32], 72);
        o[tt] = WMMA_BF16(pf, vf, o[tt]);
      }
    }

#if USE_TDM
    if (kt + 1 < 72 && threadIdx.x < 32)
      __builtin_amdgcn_s_wait_tensorcnt((short)0);  // next tile landed
#endif
    __syncthreads();
  }

  // ---- finalize: divide by l, write merged (L x 3072) bf16 ----
  float linv[8];
#pragma unroll
  for (int r = 0; r < 8; ++r) linv[r] = 1.f / sL[wave][rb + r];
  const int col = lane & 15;
#pragma unroll
  for (int tt = 0; tt < 8; ++tt)
#pragma unroll
    for (int r = 0; r < 8; ++r) {
      const size_t grow = (size_t)(qt * 64 + wr0 + rb + r);
      Oout[grow * 3072 + h * 128 + tt * 16 + col] = (__bf16)(o[tt][r] * linv[r]);
    }
}

// ---------------------------------------------------------------------------
// GEMM 2: out(f32) = attn(bf16, 4608x3072) @ {W_eproj|W_iproj}(f32) + bias.
// Row tiles 0..3 -> ctx (W_eproj), 4..35 -> img (W_iproj). img_out first.
// ---------------------------------------------------------------------------
__global__ __launch_bounds__(256)
void gemm_proj(const unsigned short* __restrict__ Aus,
               const float* __restrict__ Wi, const float* __restrict__ bi,
               const float* __restrict__ We, const float* __restrict__ be,
               float* __restrict__ Out) {
  const __bf16* A = (const __bf16*)Aus;
  const int K = 3072, N = 3072;
  __shared__ __bf16 sA[2][128][40];
  __shared__ __bf16 sBt[2][128][40];

  const int mt = blockIdx.y;
  const int m0 = mt * 128;
  const int n0 = blockIdx.x * 128;
  const int t  = threadIdx.x;
  const int wave = t >> 5;
  const int wm = (wave & 1) * 64;
  const int wn = (wave >> 1) * 32;

  const float* W;
  const float* bias;
  float* outbase;
  int orow0;
  if (mt < 4) { W = We; bias = be; outbase = Out + (size_t)4096 * 3072; orow0 = m0; }
  else        { W = Wi; bias = bi; outbase = Out;                        orow0 = m0 - 512; }

  const v8f zf = {0.f,0.f,0.f,0.f,0.f,0.f,0.f,0.f};
  v8f acc[4][2];
#pragma unroll
  for (int i = 0; i < 4; ++i)
#pragma unroll
    for (int j = 0; j < 2; ++j) acc[i][j] = zf;

  stage_a_bf16(A, sA[0], m0, K, 0, t);
  stage_bt_f32(W, sBt[0], n0, N, 0, t);
  __syncthreads();

  const int nk = K >> 5;
  for (int ks = 0; ks < nk; ++ks) {
    const int cur = ks & 1;
    if (ks + 1 < nk) {
      if (ks + 2 < nk)
        __builtin_prefetch(W + (size_t)((ks + 2) * 32) * N + n0 + (t & 127), 0, 1);
      stage_a_bf16(A, sA[cur ^ 1], m0, K, (ks + 1) * 32, t);
      stage_bt_f32(W, sBt[cur ^ 1], n0, N, (ks + 1) * 32, t);
    }
    v16bf af[4], bfr[2];
#pragma unroll
    for (int i = 0; i < 4; ++i) af[i] = frag_a_lds(&sA[cur][wm + i * 16][0], 40);
#pragma unroll
    for (int j = 0; j < 2; ++j) bfr[j] = frag_a_lds(&sBt[cur][wn + j * 16][0], 40);
#pragma unroll
    for (int i = 0; i < 4; ++i)
#pragma unroll
      for (int j = 0; j < 2; ++j)
        acc[i][j] = WMMA_BF16(af[i], bfr[j], acc[i][j]);
    __syncthreads();
  }

  const int lane = t & 31;
  const int col  = lane & 15;
  const int rb   = (lane >> 4) * 8;
#pragma unroll
  for (int i = 0; i < 4; ++i)
#pragma unroll
    for (int j = 0; j < 2; ++j) {
      const int gcol = n0 + wn + j * 16 + col;
      const float bv = bias[gcol];
#pragma unroll
      for (int r = 0; r < 8; ++r) {
        const size_t grow = (size_t)(orow0 + wm + i * 16 + rb + r);
        outbase[grow * (size_t)N + gcol] = acc[i][j][r] + bv;
      }
    }
}

// ---------------------------------------------------------------------------
extern "C" void kernel_launch(void* const* d_in, const int* in_sizes, int n_in,
                              void* d_out, int out_size, void* d_ws, size_t ws_size,
                              hipStream_t stream) {
  (void)in_sizes; (void)n_in; (void)out_size; (void)ws_size;

  const float* X_img   = (const float*)d_in[0];   // (1,4096,3072)
  const float* X_txt   = (const float*)d_in[1];   // (1,512,3072)
  const float* rot     = (const float*)d_in[2];   // (4608,64,4)
  const float* W_iqkv  = (const float*)d_in[3];   // (3072,9216)
  const float* W_eqkv  = (const float*)d_in[4];
  const float* W_iproj = (const float*)d_in[5];   // (3072,3072)
  const float* b_iproj = (const float*)d_in[6];
  const float* W_eproj = (const float*)d_in[7];
  const float* b_eproj = (const float*)d_in[8];
  const float* qn      = (const float*)d_in[9];
  const float* kn      = (const float*)d_in[10];
  const float* eqn     = (const float*)d_in[11];
  const float* ekn     = (const float*)d_in[12];
  float* out = (float*)d_out;

  // workspace (bf16 elements):
  //   [0]            qkv_raw (4608 x 9216), later reused as attn_out (4608 x 3072)
  //   [+qkvElems]    Qh (24 x 4608 x 128)
  //   [+hElems]      Kh (24 x 4608 x 128)
  //   [+hElems]      Vt (24 x 128 x 4608)   (d-major transposed V)
  unsigned short* ws0 = (unsigned short*)d_ws;
  const size_t qkvElems = (size_t)4608 * 9216;
  const size_t hElems   = (size_t)24 * 4608 * 128;
  unsigned short* Qh = ws0 + qkvElems;
  unsigned short* Kh = Qh + hElems;
  unsigned short* Vt = Kh + hElems;

  // 1) QKV GEMMs (text rows 0..511, image rows 512..4607 -> merged concat order)
  gemm_x_w_bf16out<<<dim3(72, 4), 256, 0, stream>>>(X_txt, W_eqkv, ws0, 512, 9216, 3072, 0);
  gemm_x_w_bf16out<<<dim3(72, 32), 256, 0, stream>>>(X_img, W_iqkv, ws0, 4096, 9216, 3072, 512);

  // 2) RMSNorm + RoPE (Q,K) and V transpose
  norm_rope<<<dim3(13824), 256, 0, stream>>>(ws0, rot, qn, kn, eqn, ekn, Qh, Kh);
  transpose_v<<<dim3(24, 72, 2), 256, 0, stream>>>(ws0, Vt);

  // 3) flash attention -> merged (4608 x 3072) bf16, reusing qkv_raw region
  flash_attn<<<dim3(72, 24), 128, 0, stream>>>(Qh, Kh, Vt, ws0);

  // 4) output projections -> d_out = [img_out (4096x3072) | ctx_out (512x3072)]
  gemm_proj<<<dim3(24, 36), 256, 0, stream>>>(ws0, W_iproj, b_iproj, W_eproj, b_eproj, out);
}
